// Mamba2Block_7730941133047
// MI455X (gfx1250) — matrix operand order repaired and checked by reading, not compiled, not run on hardware
//
#include <hip/hip_runtime.h>

// ---------------- problem constants ----------------
#define B_SZ 2
#define SEQ 2048
#define D_MODEL 2048
#define D_INNER 4096
#define D_STATE 128
#define NHEADS 64
#define HEADDIM 64
#define CHUNK 256
#define NC 8
#define CONV_DIM (D_INNER + 2*D_STATE)     // 4352
#define DPROJ 8512
#define DPROJ_PAD 8576                      // padded to multiple of 128
#define BL (B_SZ*SEQ)                       // 4096

typedef __bf16 v16bf __attribute__((ext_vector_type(16)));
typedef __bf16 v8bf  __attribute__((ext_vector_type(8)));
typedef float  v8f   __attribute__((ext_vector_type(8)));
typedef int    vi4   __attribute__((ext_vector_type(4)));

union FragBF { v16bf v; v8bf h[2]; };

// --------- CDNA5 async global->LDS path (guarded; falls back to sync staging) --------
#if defined(__has_builtin)
#  if __has_builtin(__builtin_amdgcn_global_load_async_to_lds_b128)
#    define HAVE_ASYNC_LDS 1
#  endif
#  if __has_builtin(__builtin_amdgcn_s_wait_asynccnt)
#    define HAVE_WAIT_ASYNC 1
#  endif
#endif
#ifndef HAVE_ASYNC_LDS
#  define HAVE_ASYNC_LDS 0
#endif
#ifndef HAVE_WAIT_ASYNC
#  define HAVE_WAIT_ASYNC 0
#endif

__device__ __forceinline__ void copy16_to_lds(const __bf16* g, __bf16* l) {
#if HAVE_ASYNC_LDS
  __builtin_amdgcn_global_load_async_to_lds_b128((vi4*)g, (vi4*)l, 0, 0);
#else
  *(v8bf*)l = *(const v8bf*)g;
#endif
}
__device__ __forceinline__ void wait_async2() {
#if HAVE_ASYNC_LDS
#  if HAVE_WAIT_ASYNC
  __builtin_amdgcn_s_wait_asynccnt(2);
#  else
  asm volatile("s_wait_asynccnt 2" ::: "memory");
#  endif
#endif
}
__device__ __forceinline__ void wait_async0() {
#if HAVE_ASYNC_LDS
#  if HAVE_WAIT_ASYNC
  __builtin_amdgcn_s_wait_asynccnt(0);
#  else
  asm volatile("s_wait_asynccnt 0" ::: "memory");
#  endif
#endif
}

__device__ __forceinline__ v8f wmma_bf16(const FragBF& a, const FragBF& b, v8f c) {
  // D = A(16x32 bf16) x B(32x16 bf16) + C(16x16 f32)
  return __builtin_amdgcn_wmma_f32_16x16x32_bf16(false, a.v, false, b.v, (short)0, c, false, false);
}

__device__ __forceinline__ float softplusf(float x) {
  return (x > 20.f) ? x : __logf(1.f + __expf(x));
}
__device__ __forceinline__ float siluf(float x) { return x / (1.f + __expf(-x)); }

// ---------------- fp32 -> bf16 converts ----------------
__global__ void cvt_kernel(const float* __restrict__ src, __bf16* __restrict__ dst, long n) {
  long i = (long)blockIdx.x * 256 + threadIdx.x;
  if (i < n) dst[i] = (__bf16)src[i];
}

__global__ void cvt_pad_rows_kernel(const float* __restrict__ src, __bf16* __restrict__ dst,
                                    int src_rows, int dst_rows, int cols) {
  long i = (long)blockIdx.x * 256 + threadIdx.x;
  long n = (long)dst_rows * cols;
  if (i >= n) return;
  long r = i / cols, c = i % cols;
  dst[i] = (r < src_rows) ? (__bf16)src[r * cols + c] : (__bf16)0.0f;
}

// ---------------- generic GEMM: C[M,N] = A[M,K] * W[N,K]^T ----------------
// block 256 thr = 8 waves; block tile 256(M) x 128(N); wave tile 32x128.
// W K-slab (128 x 32 bf16) double-buffered in LDS via async global->LDS.
#define WT_STRIDE 40  // 32 + 8 pad (keeps 16B alignment: 40*2=80B)
template<bool OUT_BF16>
__global__ __launch_bounds__(256)
void gemm_abt(const __bf16* __restrict__ A, const __bf16* __restrict__ W,
              void* __restrict__ Cout, int K, int lda, int ldb, int ldc) {
  __shared__ __bf16 Wt[2][128][WT_STRIDE];
  const int t = threadIdx.x;
  const int w = t >> 5, lane = t & 31, lr = lane & 15, kh = lane >> 4;
  const long m0 = (long)blockIdx.y * 256 + w * 32;
  const long n0 = (long)blockIdx.x * 128;

  v8f acc[2][8];
  for (int mt = 0; mt < 2; ++mt)
    for (int nt = 0; nt < 8; ++nt)
      for (int i = 0; i < 8; ++i) acc[mt][nt][i] = 0.f;

  // stage W slab [128 rows x 32 K] : 512 x 16B chunks, 2 per thread
  auto stage = [&](int buf, int k0) {
#pragma unroll
    for (int j = 0; j < 2; ++j) {
      int cid = t * 2 + j;          // 0..511
      int row = cid >> 2;           // 0..127
      int part = cid & 3;           // 16B quarter of the 64B row slab
      copy16_to_lds(W + (n0 + row) * (long)ldb + k0 + part * 8,
                    &Wt[buf][row][part * 8]);
    }
  };

  const int nk = K >> 5;
  stage(0, 0);
  for (int ks = 0; ks < nk; ++ks) {
    const int buf = ks & 1;
    __syncthreads();                       // everyone done reading buf^1 (iter ks-1)
    if (ks + 1 < nk) { stage(buf ^ 1, (ks + 1) * 32); wait_async2(); }
    else             { wait_async0(); }
    __syncthreads();                       // staged slab visible block-wide
    const int k0 = ks * 32;

    FragBF af[2];
#pragma unroll
    for (int mt = 0; mt < 2; ++mt) {
      const __bf16* p = A + (m0 + mt * 16 + lr) * (long)lda + k0 + kh * 8;
      af[mt].h[0] = *(const v8bf*)p;
      af[mt].h[1] = *(const v8bf*)(p + 16);
    }
#pragma unroll
    for (int nt = 0; nt < 8; ++nt) {
      FragBF bfm;
      const __bf16* q = &Wt[buf][nt * 16 + lr][kh * 8];
      bfm.h[0] = *(const v8bf*)q;
      bfm.h[1] = *(const v8bf*)(q + 16);
      acc[0][nt] = wmma_bf16(af[0], bfm, acc[0][nt]);
      acc[1][nt] = wmma_bf16(af[1], bfm, acc[1][nt]);
    }
  }
#pragma unroll
  for (int mt = 0; mt < 2; ++mt)
#pragma unroll
    for (int nt = 0; nt < 8; ++nt)
#pragma unroll
      for (int i = 0; i < 8; ++i) {
        long row = m0 + mt * 16 + i + 8 * kh;
        long col = n0 + nt * 16 + lr;
        if (OUT_BF16) ((__bf16*)Cout)[row * (long)ldc + col] = (__bf16)acc[mt][nt][i];
        else          ((float*)Cout)[row * (long)ldc + col] = acc[mt][nt][i];
      }
}

// ---------------- depthwise causal conv (w=4) + bias + silu + dt-scale ----------------
__global__ void conv_kernel(const __bf16* __restrict__ zx, const float* __restrict__ conv_w,
                            const float* __restrict__ conv_b, const float* __restrict__ dt_bias,
                            __bf16* __restrict__ xsb, __bf16* __restrict__ Xb,
                            __bf16* __restrict__ Bbm, __bf16* __restrict__ Cbm) {
  long i = (long)blockIdx.x * 256 + threadIdx.x;
  long total = (long)BL * CONV_DIM;
  if (i >= total) return;
  int ch = (int)(i % CONV_DIM);
  long row = i / CONV_DIM;                // b*SEQ + l
  int b = (int)(row >> 11), l = (int)(row & 2047);
  float acc = conv_b[ch];
#pragma unroll
  for (int j = 0; j < 4; ++j) {
    int ll = l - 3 + j;
    if (ll >= 0)
      acc += conv_w[ch * 4 + j] * (float)zx[((long)(b << 11) + ll) * DPROJ_PAD + D_INNER + ch];
  }
  float s = siluf(acc);
  if (ch < D_INNER) {
    int h = ch >> 6;
    float raw = (float)zx[row * DPROJ_PAD + D_INNER + CONV_DIM + h] + dt_bias[h];
    float dt = softplusf(raw);
    xsb[row * D_INNER + ch] = (__bf16)s;
    Xb[row * D_INNER + ch]  = (__bf16)(s * dt);
  } else if (ch < D_INNER + D_STATE) {
    Bbm[row * D_STATE + (ch - D_INNER)] = (__bf16)s;
  } else {
    Cbm[row * D_STATE + (ch - D_INNER - D_STATE)] = (__bf16)s;
  }
}

// ---------------- dA[b][h][l] = softplus(dt_raw+bias) * (-exp(A_log)) ----------------
__global__ void dA_kernel(const __bf16* __restrict__ zx, const float* __restrict__ dt_bias,
                          const float* __restrict__ A_log, float* __restrict__ dA) {
  long i = (long)blockIdx.x * 256 + threadIdx.x;
  if (i >= (long)BL * NHEADS) return;
  int h = (int)(i & 63);
  long bl = i >> 6;
  float raw = (float)zx[bl * DPROJ_PAD + D_INNER + CONV_DIM + h] + dt_bias[h];
  float dt = softplusf(raw);
  long b = bl >> 11, l = bl & 2047;
  dA[((b * NHEADS) + h) * (long)SEQ + l] = dt * (-__expf(A_log[h]));
}

// ---------------- per-chunk inclusive cumsum of dA ----------------
__global__ void cumsum_kernel(const float* __restrict__ dA, float* __restrict__ Acs,
                              float* __restrict__ csum) {
  int blk = blockIdx.x;          // (b*64+h)*8 + c
  int c = blk & 7, bh = blk >> 3;
  __shared__ float s[CHUNK];
  int t = threadIdx.x;
  s[t] = dA[(long)bh * SEQ + c * CHUNK + t];
  __syncthreads();
  for (int off = 1; off < CHUNK; off <<= 1) {
    float add = (t >= off) ? s[t - off] : 0.f;
    __syncthreads();
    s[t] += add;
    __syncthreads();
  }
  Acs[(long)blk * CHUNK + t] = s[t];
  if (t == 255) csum[blk] = s[255];
}

// ---------------- G[b,c] = Cc(256x128) * Bc(256x128)^T  (shared across heads) ----------
__global__ __launch_bounds__(256)
void g_kernel(const __bf16* __restrict__ Bbm, const __bf16* __restrict__ Cbm,
              float* __restrict__ Gf) {
  int blk = blockIdx.x;          // b*NC + c
  int t = threadIdx.x;
  int w = t >> 5, lane = t & 31, lr = lane & 15, kh = lane >> 4;
  long rowbase = (long)blk * CHUNK;
  const __bf16* Crow = Cbm + rowbase * D_STATE;
  const __bf16* Brow = Bbm + rowbase * D_STATE;
  float* Gout = Gf + (long)blk * CHUNK * CHUNK;
  int m0 = w * 32;
  FragBF a[2][4];
#pragma unroll
  for (int mt = 0; mt < 2; ++mt) {
    int row = m0 + mt * 16 + lr;
#pragma unroll
    for (int ks = 0; ks < 4; ++ks) {
      const __bf16* p = Crow + (long)row * D_STATE + ks * 32 + kh * 8;
      a[mt][ks].h[0] = *(const v8bf*)p;
      a[mt][ks].h[1] = *(const v8bf*)(p + 16);
    }
  }
  for (int nt = 0; nt < 16; ++nt) {
    v8f acc0, acc1;
    for (int i = 0; i < 8; ++i) { acc0[i] = 0.f; acc1[i] = 0.f; }
    int col = nt * 16 + lr;
#pragma unroll
    for (int ks = 0; ks < 4; ++ks) {
      FragBF bfm;
      const __bf16* p = Brow + (long)col * D_STATE + ks * 32 + kh * 8;
      bfm.h[0] = *(const v8bf*)p;
      bfm.h[1] = *(const v8bf*)(p + 16);
      acc0 = wmma_bf16(a[0][ks], bfm, acc0);
      acc1 = wmma_bf16(a[1][ks], bfm, acc1);
    }
#pragma unroll
    for (int i = 0; i < 8; ++i) {
      Gout[(long)(m0 + i + 8 * kh) * CHUNK + nt * 16 + lr]       = acc0[i];
      Gout[(long)(m0 + 16 + i + 8 * kh) * CHUNK + nt * 16 + lr]  = acc1[i];
    }
  }
}

// ---------------- states[b,c,h][n][p] = sum_l Bc[l,n] * decay[l] * X[l,p] ------------
__global__ __launch_bounds__(256)
void states_kernel(const __bf16* __restrict__ Bbm, const __bf16* __restrict__ Xb,
                   const float* __restrict__ Acs, float* __restrict__ statesf) {
  int blk = blockIdx.x;          // (b*NC+c)*64 + h
  int h = blk & 63, bc = blk >> 6;
  int b = bc >> 3, c = bc & 7;
  __shared__ __bf16 BcT[D_STATE][CHUNK + 8];
  __shared__ __bf16 XdT[HEADDIM][CHUNK + 8];
  __shared__ float dec[CHUNK];
  int t = threadIdx.x;
  long rowbase = (long)bc * CHUNK;
  const float* acs = Acs + ((long)(b * NHEADS + h) * NC + c) * CHUNK;
  float alast = acs[CHUNK - 1];
  dec[t] = __expf(alast - acs[t]);
  __syncthreads();
  for (int i = 0; i < 128; ++i) {
    int e = t + i * 256;
    int l = e >> 7, n = e & 127;
    BcT[n][l] = Bbm[(rowbase + l) * D_STATE + n];
  }
  for (int i = 0; i < 64; ++i) {
    int e = t + i * 256;
    int l = e >> 6, p = e & 63;
    XdT[p][l] = (__bf16)((float)Xb[(rowbase + l) * D_INNER + h * HEADDIM + p] * dec[l]);
  }
  __syncthreads();
  int w = t >> 5, lane = t & 31, lr = lane & 15, kh = lane >> 4;
  v8f acc[4];
  for (int nt = 0; nt < 4; ++nt)
    for (int i = 0; i < 8; ++i) acc[nt][i] = 0.f;
  for (int ks = 0; ks < 8; ++ks) {
    int k0 = ks * 32;
    FragBF af;
    const __bf16* p = &BcT[w * 16 + lr][k0 + kh * 8];
    af.h[0] = *(const v8bf*)p;
    af.h[1] = *(const v8bf*)(p + 16);
#pragma unroll
    for (int nt = 0; nt < 4; ++nt) {
      FragBF bfm;
      const __bf16* q = &XdT[nt * 16 + lr][k0 + kh * 8];
      bfm.h[0] = *(const v8bf*)q;
      bfm.h[1] = *(const v8bf*)(q + 16);
      acc[nt] = wmma_bf16(af, bfm, acc[nt]);
    }
  }
  float* sout = statesf + ((long)bc * NHEADS + h) * (D_STATE * HEADDIM);
#pragma unroll
  for (int nt = 0; nt < 4; ++nt)
#pragma unroll
    for (int i = 0; i < 8; ++i) {
      int n = w * 16 + i + 8 * kh;
      int p = nt * 16 + lr;
      sout[(long)n * HEADDIM + p] = acc[nt][i];
    }
}

// ---------------- inter-chunk scan: prev[c] = P; P = exp(s_c)*P + states[c] ----------
__global__ void scan_kernel(const float* __restrict__ statesf, const float* __restrict__ csum,
                            __bf16* __restrict__ prevb) {
  int bh = blockIdx.x;           // b*64 + h
  int b = bh >> 6, h = bh & 63;
  int t = threadIdx.x;
  float P[32];
#pragma unroll
  for (int j = 0; j < 32; ++j) P[j] = 0.f;
  for (int c = 0; c < NC; ++c) {
    long sbase = ((long)(b * NC + c) * NHEADS + h) * (D_STATE * HEADDIM);
    float mult = __expf(csum[(long)bh * NC + c]);
#pragma unroll
    for (int j = 0; j < 32; ++j) {
      int e = j * 256 + t;
      int n = e >> 6, p = e & 63;
      prevb[sbase + (long)p * D_STATE + n] = (__bf16)P[j];
      P[j] = mult * P[j] + statesf[sbase + e];
    }
  }
}

// ---------------- Y = (G.*L)X + exp(Acs)*(C prev^T) + D*x_ssm  ------------------------
__global__ __launch_bounds__(256)
void y_kernel(const float* __restrict__ Gf, const __bf16* __restrict__ Cbm,
              const __bf16* __restrict__ prevb, const __bf16* __restrict__ Xb,
              const __bf16* __restrict__ xsb, const float* __restrict__ Acs,
              const float* __restrict__ Dv, __bf16* __restrict__ Yb) {
  int blk = blockIdx.x;          // (b*NC+c)*64 + h
  int h = blk & 63, bc = blk >> 6;
  int b = bc >> 3, c = bc & 7;
  __shared__ __bf16 XT[HEADDIM][CHUNK + 8];
  __shared__ float eA[CHUNK], enA[CHUNK];
  int t = threadIdx.x;
  long rowbase = (long)bc * CHUNK;
  for (int i = 0; i < 64; ++i) {
    int e = t + i * 256;
    int l = e >> 6, p = e & 63;
    XT[p][l] = Xb[(rowbase + l) * D_INNER + h * HEADDIM + p];
  }
  {
    float a = Acs[((long)(b * NHEADS + h) * NC + c) * CHUNK + t];
    eA[t] = __expf(a);
    enA[t] = __expf(-a);
  }
  __syncthreads();
  int w = t >> 5, lane = t & 31, lr = lane & 15, kh = lane >> 4;
  int m0 = w * 32;
  v8f acc[2][4];
  for (int mt = 0; mt < 2; ++mt)
    for (int nt = 0; nt < 4; ++nt)
      for (int i = 0; i < 8; ++i) acc[mt][nt][i] = 0.f;

  // --- Y_off: A = Cc (256x128), B = prev^T (prevb stored [p][n]) ---
  const __bf16* prevh = prevb + ((long)bc * NHEADS + h) * (D_STATE * HEADDIM);
  for (int ks = 0; ks < 4; ++ks) {
    int k0 = ks * 32;
    FragBF af[2];
#pragma unroll
    for (int mt = 0; mt < 2; ++mt) {
      const __bf16* p = Cbm + (rowbase + m0 + mt * 16 + lr) * D_STATE + k0 + kh * 8;
      af[mt].h[0] = *(const v8bf*)p;
      af[mt].h[1] = *(const v8bf*)(p + 16);
    }
#pragma unroll
    for (int nt = 0; nt < 4; ++nt) {
      FragBF bfm;
      const __bf16* q = prevh + (long)(nt * 16 + lr) * D_STATE + k0 + kh * 8;
      bfm.h[0] = *(const v8bf*)q;
      bfm.h[1] = *(const v8bf*)(q + 16);
      acc[0][nt] = wmma_bf16(af[0], bfm, acc[0][nt]);
      acc[1][nt] = wmma_bf16(af[1], bfm, acc[1][nt]);
    }
  }
#pragma unroll
  for (int mt = 0; mt < 2; ++mt)
#pragma unroll
    for (int nt = 0; nt < 4; ++nt)
#pragma unroll
      for (int i = 0; i < 8; ++i)
        acc[mt][nt][i] *= eA[m0 + mt * 16 + i + 8 * kh];

  // --- Y_diag: A = G .* L (masked, scaled), B = X^T from LDS ---
  const float* Gc = Gf + (long)bc * (CHUNK * CHUNK);
  for (int ks = 0; ks < 8; ++ks) {
    int k0 = ks * 32;
    FragBF af[2];
#pragma unroll
    for (int mt = 0; mt < 2; ++mt) {
      int m = m0 + mt * 16 + lr;
      const float* gp = Gc + (long)m * CHUNK;
      float em = eA[m];
#pragma unroll
      for (int j = 0; j < 8; ++j) {
        int s1 = k0 + kh * 8 + j;
        int s2 = s1 + 16;
        float v1 = (m >= s1) ? gp[s1] * em * enA[s1] : 0.f;
        float v2 = (m >= s2) ? gp[s2] * em * enA[s2] : 0.f;
        af[mt].v[j] = (__bf16)v1;
        af[mt].v[8 + j] = (__bf16)v2;
      }
    }
#pragma unroll
    for (int nt = 0; nt < 4; ++nt) {
      FragBF bfm;
      const __bf16* q = &XT[nt * 16 + lr][k0 + kh * 8];
      bfm.h[0] = *(const v8bf*)q;
      bfm.h[1] = *(const v8bf*)(q + 16);
      acc[0][nt] = wmma_bf16(af[0], bfm, acc[0][nt]);
      acc[1][nt] = wmma_bf16(af[1], bfm, acc[1][nt]);
    }
  }
  // --- skip connection + store ---
  float Dh = Dv[h];
#pragma unroll
  for (int mt = 0; mt < 2; ++mt)
#pragma unroll
    for (int nt = 0; nt < 4; ++nt)
#pragma unroll
      for (int i = 0; i < 8; ++i) {
        int m = m0 + mt * 16 + i + 8 * kh;
        int p = nt * 16 + lr;
        long gi = (rowbase + m) * (long)D_INNER + h * HEADDIM + p;
        float v = acc[mt][nt][i] + Dh * (float)xsb[gi];
        Yb[gi] = (__bf16)v;
      }
}

// ---------------- gate (silu(z)) + RMS norm over D_INNER ----------------
__global__ void gate_norm_kernel(const __bf16* __restrict__ zx, const __bf16* __restrict__ Yb,
                                 const float* __restrict__ norm_w, __bf16* __restrict__ ynb) {
  int row = blockIdx.x;
  int t = threadIdx.x;
  float yg[16];
  float ss = 0.f;
#pragma unroll
  for (int i = 0; i < 16; ++i) {
    int col = t + i * 256;
    float z = (float)zx[(long)row * DPROJ_PAD + col];
    float y = (float)Yb[(long)row * D_INNER + col];
    float g = y * siluf(z);
    yg[i] = g;
    ss += g * g;
  }
  __shared__ float red[256];
  red[t] = ss;
  __syncthreads();
  for (int off = 128; off > 0; off >>= 1) {
    if (t < off) red[t] += red[t + off];
    __syncthreads();
  }
  float scale = rsqrtf(red[0] / (float)D_INNER + 1e-5f);
#pragma unroll
  for (int i = 0; i < 16; ++i) {
    int col = t + i * 256;
    ynb[(long)row * D_INNER + col] = (__bf16)(yg[i] * scale * norm_w[col]);
  }
}

// =====================================================================================
extern "C" void kernel_launch(void* const* d_in, const int* in_sizes, int n_in,
                              void* d_out, int out_size, void* d_ws, size_t ws_size,
                              hipStream_t stream) {
  const float* x          = (const float*)d_in[0];
  const float* in_proj_w  = (const float*)d_in[1];
  const float* conv_w     = (const float*)d_in[2];
  const float* conv_b     = (const float*)d_in[3];
  const float* dt_bias    = (const float*)d_in[4];
  const float* A_log      = (const float*)d_in[5];
  const float* Dv         = (const float*)d_in[6];
  const float* norm_w     = (const float*)d_in[7];
  const float* out_proj_w = (const float*)d_in[8];
  float* out = (float*)d_out;

  char* ws = (char*)d_ws;
  size_t off = 0;
  auto alloc = [&](size_t bytes) -> void* {
    off = (off + 255) & ~(size_t)255;
    void* p = ws + off;
    off += bytes;
    return p;
  };

  __bf16* xb      = (__bf16*)alloc((size_t)BL * D_MODEL * 2);
  __bf16* Wb      = (__bf16*)alloc((size_t)DPROJ_PAD * D_MODEL * 2);
  __bf16* Wob     = (__bf16*)alloc((size_t)D_MODEL * D_INNER * 2);
  __bf16* zx      = (__bf16*)alloc((size_t)BL * DPROJ_PAD * 2);
  __bf16* xsb     = (__bf16*)alloc((size_t)BL * D_INNER * 2);
  __bf16* Xb      = (__bf16*)alloc((size_t)BL * D_INNER * 2);
  __bf16* Bbm     = (__bf16*)alloc((size_t)BL * D_STATE * 2);
  __bf16* Cbm     = (__bf16*)alloc((size_t)BL * D_STATE * 2);
  float*  dA      = (float*)alloc((size_t)B_SZ * NHEADS * SEQ * 4);
  float*  Acs     = (float*)alloc((size_t)B_SZ * NHEADS * NC * CHUNK * 4);
  float*  csum    = (float*)alloc((size_t)B_SZ * NHEADS * NC * 4);
  float*  Gf      = (float*)alloc((size_t)B_SZ * NC * CHUNK * CHUNK * 4);
  float*  statesf = (float*)alloc((size_t)B_SZ * NC * NHEADS * D_STATE * HEADDIM * 4);
  __bf16* prevb   = (__bf16*)alloc((size_t)B_SZ * NC * NHEADS * D_STATE * HEADDIM * 2);
  __bf16* Yb      = (__bf16*)alloc((size_t)BL * D_INNER * 2);
  __bf16* ynb     = (__bf16*)alloc((size_t)BL * D_INNER * 2);

  {
    long n = (long)BL * D_MODEL;
    cvt_kernel<<<(n + 255) / 256, 256, 0, stream>>>(x, xb, n);
  }
  {
    long n = (long)DPROJ_PAD * D_MODEL;
    cvt_pad_rows_kernel<<<(n + 255) / 256, 256, 0, stream>>>(in_proj_w, Wb, DPROJ, DPROJ_PAD, D_MODEL);
  }
  {
    long n = (long)D_MODEL * D_INNER;
    cvt_kernel<<<(n + 255) / 256, 256, 0, stream>>>(out_proj_w, Wob, n);
  }

  // in_proj GEMM: zx[BL, DPROJ_PAD] = xb * Wb^T   (bf16 out)
  gemm_abt<true><<<dim3(DPROJ_PAD / 128, BL / 256), 256, 0, stream>>>(
      xb, Wb, zx, D_MODEL, D_MODEL, D_MODEL, DPROJ_PAD);

  {
    long n = (long)BL * CONV_DIM;
    conv_kernel<<<(n + 255) / 256, 256, 0, stream>>>(zx, conv_w, conv_b, dt_bias,
                                                     xsb, Xb, Bbm, Cbm);
  }
  {
    long n = (long)BL * NHEADS;
    dA_kernel<<<(n + 255) / 256, 256, 0, stream>>>(zx, dt_bias, A_log, dA);
  }
  cumsum_kernel<<<B_SZ * NHEADS * NC, 256, 0, stream>>>(dA, Acs, csum);
  g_kernel<<<B_SZ * NC, 256, 0, stream>>>(Bbm, Cbm, Gf);
  states_kernel<<<B_SZ * NC * NHEADS, 256, 0, stream>>>(Bbm, Xb, Acs, statesf);
  scan_kernel<<<B_SZ * NHEADS, 256, 0, stream>>>(statesf, csum, prevb);
  y_kernel<<<B_SZ * NC * NHEADS, 256, 0, stream>>>(Gf, Cbm, prevb, Xb, xsb, Acs, Dv, Yb);
  gate_norm_kernel<<<BL, 256, 0, stream>>>(zx, Yb, norm_w, ynb);

  // out_proj GEMM: out[BL, D_MODEL] = ynb * Wob^T  (f32 out)
  gemm_abt<false><<<dim3(D_MODEL / 128, BL / 256), 256, 0, stream>>>(
      ynb, Wob, out, D_INNER, D_INNER, D_INNER, D_MODEL);
}
